// NeuralDNAModel_27676769255882
// MI455X (gfx1250) — matrix operand order repaired and checked by reading, not compiled, not run on hardware
//
#include <hip/hip_runtime.h>
#include <stdint.h>
#include <stddef.h>

// ---------------- model dims ----------------
#define NTOK   2048        // B*T
#define DD     1024        // D
#define GG     16384       // G
#define GDIM   64          // GD
#define KTOP   32
#define LLAY   4
#define VV     32000
#define GEMM_K 1024        // both big GEMMs contract over D
#define RMS_EPS 1.1920929e-07f

typedef __attribute__((ext_vector_type(16))) __bf16 v16bf;
typedef __attribute__((ext_vector_type(8)))  __bf16 v8bf;
typedef __attribute__((ext_vector_type(8)))  float  v8f;
typedef __attribute__((ext_vector_type(4)))  int    v4i;

// explicit global-address-space types (guarantee global_load/global_store, not flat)
typedef __attribute__((address_space(1))) __bf16 g_bf;
typedef __attribute__((address_space(1))) v8bf   g_v8bf;
typedef __attribute__((address_space(1))) v16bf  g_v16bf;
typedef __attribute__((address_space(1))) float  g_float;

#if defined(__has_builtin)
#if __has_builtin(__builtin_amdgcn_global_load_async_to_lds_b128) && \
    __has_builtin(__builtin_amdgcn_s_wait_asynccnt)
#define HAVE_ASYNC_LDS 1
#endif
#endif

__device__ __forceinline__ unsigned short f2bf(float f) {
  unsigned u = __float_as_uint(f);
  u += 0x7FFFu + ((u >> 16) & 1u);       // round-to-nearest-even
  return (unsigned short)(u >> 16);
}

__device__ __forceinline__ v8f zero_v8f() {
  v8f z;
#pragma unroll
  for (int i = 0; i < 8; ++i) z[i] = 0.0f;
  return z;
}

// ---------------- embedding gather: x[t,:] = embed[token_ids[t],:] ----------------
__global__ __launch_bounds__(256) void k_embed(const int* __restrict__ tok,
                                               const float* __restrict__ embed,
                                               float* __restrict__ x) {
  int t = blockIdx.x;
  int id = tok[t];
  const float4* s = (const float4*)(embed + (size_t)id * DD);
  float4* d = (float4*)(x + (size_t)t * DD);
  d[threadIdx.x] = s[threadIdx.x];
}

// ---------------- RMSNorm (+ bf16 cast of the normed activations) ----------------
__global__ __launch_bounds__(256) void k_rmsnorm(const float* __restrict__ x,
                                                 const float* __restrict__ w,
                                                 float* __restrict__ xn,
                                                 unsigned short* __restrict__ xnb) {
  int t = blockIdx.x;
  int tid = threadIdx.x;
  float4 v = ((const float4*)(x + (size_t)t * DD))[tid];
  float ss = v.x * v.x + v.y * v.y + v.z * v.z + v.w * v.w;
#pragma unroll
  for (int off = 16; off; off >>= 1) ss += __shfl_down(ss, off);
  __shared__ float sred[8];
  int lane = tid & 31, wv = tid >> 5;
  if (lane == 0) sred[wv] = ss;
  __syncthreads();
  float tot = 0.0f;
#pragma unroll
  for (int i = 0; i < 8; ++i) tot += sred[i];
  float inv = rsqrtf(tot * (1.0f / (float)DD) + RMS_EPS);
  float4 wv4 = ((const float4*)w)[tid];
  float4 o;
  o.x = v.x * inv * wv4.x;
  o.y = v.y * inv * wv4.y;
  o.z = v.z * inv * wv4.z;
  o.w = v.w * inv * wv4.w;
  ((float4*)(xn + (size_t)t * DD))[tid] = o;
  ushort4 ob;
  ob.x = f2bf(o.x); ob.y = f2bf(o.y); ob.z = f2bf(o.z); ob.w = f2bf(o.w);
  ((ushort4*)(xnb + (size_t)t * DD))[tid] = ob;
}

// ---------------- fp32 -> bf16 weight cast ----------------
__global__ __launch_bounds__(256) void k_cast_bf16(const float* __restrict__ src,
                                                   unsigned short* __restrict__ dst,
                                                   int n4) {
  int i = blockIdx.x * 256 + threadIdx.x;
  if (i >= n4) return;
  float4 v = ((const float4*)src)[i];
  ushort4 o;
  o.x = f2bf(v.x); o.y = f2bf(v.y); o.z = f2bf(v.z); o.w = f2bf(v.w);
  ((ushort4*)dst)[i] = o;
}

// ---------------- bf16 WMMA GEMM: C[M,N] = alpha * A[M,K] * B[N,K]^T, K = GEMM_K ------
// block = 256 threads = 8 waves; block tile 64(M) x 256(N); wave tile 32x64.
// 8 WMMAs per 12 x global_load_b128 per k-step per wave.  grid = (N/256, M/64)
__global__ __launch_bounds__(256) void k_gemm_bf16(const unsigned short* Au,
                                                   const unsigned short* Bu,
                                                   float* Cp,
                                                   int N,
                                                   const float* tptr) {
  g_bf* A = (g_bf*)(void*)Au;
  g_bf* Bm = (g_bf*)(void*)Bu;
  g_float* C = (g_float*)(void*)Cp;

  int lane = threadIdx.x & 31;
  int wave = threadIdx.x >> 5;
  int wm = wave >> 2;            // 0..1
  int wn = wave & 3;             // 0..3
  int m0 = blockIdx.y * 64 + wm * 32;
  int n0 = blockIdx.x * 256 + wn * 64;
  int r = lane & 15, h = lane >> 4;

  v8f acc00 = zero_v8f(), acc01 = zero_v8f(), acc02 = zero_v8f(), acc03 = zero_v8f();
  v8f acc10 = zero_v8f(), acc11 = zero_v8f(), acc12 = zero_v8f(), acc13 = zero_v8f();

  // A fragment: lane holds row (m0+r); K elems h*16 .. h*16+15 (contiguous)
  g_bf* aRow = A + (size_t)(m0 + r) * GEMM_K + h * 16;
  // B fragment: lane holds W-row (n0+r); K elems h*16 .. h*16+15 (contiguous)
  g_bf* bRow = Bm + (size_t)(n0 + r) * GEMM_K + h * 16;

  for (int k0 = 0; k0 < GEMM_K; k0 += 32) {
    v8bf a0lo = *(g_v8bf*)(aRow);
    v8bf a0hi = *(g_v8bf*)(aRow + 8);
    v8bf a1lo = *(g_v8bf*)(aRow + 16 * GEMM_K);
    v8bf a1hi = *(g_v8bf*)(aRow + 16 * GEMM_K + 8);
    v16bf fa0 = __builtin_shufflevector(a0lo, a0hi, 0,1,2,3,4,5,6,7,8,9,10,11,12,13,14,15);
    v16bf fa1 = __builtin_shufflevector(a1lo, a1hi, 0,1,2,3,4,5,6,7,8,9,10,11,12,13,14,15);
    v16bf fb0 = *(g_v16bf*)(bRow);
    v16bf fb1 = *(g_v16bf*)(bRow + 16 * GEMM_K);
    v16bf fb2 = *(g_v16bf*)(bRow + 32 * GEMM_K);
    v16bf fb3 = *(g_v16bf*)(bRow + 48 * GEMM_K);
    acc00 = __builtin_amdgcn_wmma_f32_16x16x32_bf16(false, fa0, false, fb0, (short)0, acc00, false, false);
    acc01 = __builtin_amdgcn_wmma_f32_16x16x32_bf16(false, fa0, false, fb1, (short)0, acc01, false, false);
    acc02 = __builtin_amdgcn_wmma_f32_16x16x32_bf16(false, fa0, false, fb2, (short)0, acc02, false, false);
    acc03 = __builtin_amdgcn_wmma_f32_16x16x32_bf16(false, fa0, false, fb3, (short)0, acc03, false, false);
    acc10 = __builtin_amdgcn_wmma_f32_16x16x32_bf16(false, fa1, false, fb0, (short)0, acc10, false, false);
    acc11 = __builtin_amdgcn_wmma_f32_16x16x32_bf16(false, fa1, false, fb1, (short)0, acc11, false, false);
    acc12 = __builtin_amdgcn_wmma_f32_16x16x32_bf16(false, fa1, false, fb2, (short)0, acc12, false, false);
    acc13 = __builtin_amdgcn_wmma_f32_16x16x32_bf16(false, fa1, false, fb3, (short)0, acc13, false, false);
    aRow += 32;
    bRow += 32;
  }

  float alpha = tptr ? (1.0f / fmaxf(tptr[0], 0.1f)) : 1.0f;

  // C layout: VGPR q, lanes 0-15 -> row m0+q col n0+lane; lanes 16-31 -> row m0+8+q
  g_float* cbase = C + (size_t)(m0 + h * 8) * N + (n0 + r);
#pragma unroll
  for (int q = 0; q < 8; ++q) {
    g_float* cp0 = cbase + (size_t)q * N;
    cp0[0]  = acc00[q] * alpha;
    cp0[16] = acc01[q] * alpha;
    cp0[32] = acc02[q] * alpha;
    cp0[48] = acc03[q] * alpha;
    g_float* cp1 = cp0 + (size_t)16 * N;
    cp1[0]  = acc10[q] * alpha;
    cp1[16] = acc11[q] * alpha;
    cp1[32] = acc12[q] * alpha;
    cp1[48] = acc13[q] * alpha;
  }
}

// ---------------- top-32 per token (one wave per token, row staged in LDS) ----------------
__global__ __launch_bounds__(32) void k_topk(const float* __restrict__ logits,
                                             float* __restrict__ tv,
                                             int* __restrict__ ti) {
  __shared__ float sv[GG];                          // 64 KB (CDNA5 WGP has 320 KB)
  int t = blockIdx.x;
  int lane = threadIdx.x;

#if defined(HAVE_ASYNC_LDS)
  {
    // DMA straight to LDS: GLOBAL_LOAD_ASYNC_TO_LDS_B128, tracked by ASYNCcnt.
    typedef __attribute__((address_space(1))) v4i* gp_v4i;
    typedef __attribute__((address_space(3))) v4i* lp_v4i;
    gp_v4i gs = (gp_v4i)(void*)(logits + (size_t)t * GG);
    lp_v4i ls = (lp_v4i)(void*)sv;
    for (int i = lane; i < GG / 4; i += 32)
      __builtin_amdgcn_global_load_async_to_lds_b128(gs + i, ls + i, 0, 0);
    __builtin_amdgcn_s_wait_asynccnt(0);
  }
#else
  {
    const float4* src = (const float4*)(logits + (size_t)t * GG);
    float4* dst = (float4*)sv;
    for (int i = lane; i < GG / 4; i += 32) dst[i] = src[i];
  }
#endif
  __syncthreads();

  float* tvp = tv + (size_t)t * KTOP;
  int* tip = ti + (size_t)t * KTOP;

  for (int kk = 0; kk < KTOP; ++kk) {
    float bv = -3.402823466e38f;
    int bi = 0x7FFFFFFF;
    for (int i = lane; i < GG; i += 32) {
      float f = sv[i];
      if (f > bv || (f == bv && i < bi)) { bv = f; bi = i; }
    }
#pragma unroll
    for (int off = 16; off; off >>= 1) {
      float ov = __shfl_down(bv, off);
      int oi = __shfl_down(bi, off);
      if (ov > bv || (ov == bv && oi < bi)) { bv = ov; bi = oi; }
    }
    bv = __shfl(bv, 0);
    bi = __shfl(bi, 0);
    if (lane == 0) {
      tvp[kk] = bv;
      tip[kk] = bi;
      sv[bi] = -__builtin_inff();
    }
    __syncthreads();
  }
}

// ---------------- softmax over top-k + gene mixing (8 tokens per block, wave per token) ----
__global__ __launch_bounds__(256) void k_express(const float* __restrict__ tv,
                                                 const int* __restrict__ ti,
                                                 const float* __restrict__ genes,
                                                 float* __restrict__ expr) {
  int lane = threadIdx.x & 31;
  int wv = threadIdx.x >> 5;
  int t = blockIdx.x * 8 + wv;
  float v = tv[(size_t)t * KTOP + lane];
  int id = ti[(size_t)t * KTOP + lane];
  float m = __shfl(v, 0);                 // values sorted descending
  float e = expf(v - m);
  float s = e;
#pragma unroll
  for (int off = 16; off; off >>= 1) s += __shfl_down(s, off);
  s = __shfl(s, 0);
  float p = e / s;
  float a0 = 0.0f, a1 = 0.0f;
#pragma unroll 4
  for (int k = 0; k < KTOP; ++k) {
    float pk = __shfl(p, k);
    int ik = __shfl(id, k);
    const float* g = genes + (size_t)ik * GDIM;
    a0 += pk * g[lane];
    a1 += pk * g[lane + 32];
  }
  expr[(size_t)t * GDIM + lane] = a0;
  expr[(size_t)t * GDIM + lane + 32] = a1;
}

// ---------------- down/up projections + gated residual update ----------------
__global__ __launch_bounds__(256) void k_ffn(float* __restrict__ x,
                                             const float* __restrict__ xn,
                                             const float* __restrict__ expr,
                                             const float* __restrict__ g2d,
                                             const float* __restrict__ g2u,
                                             const float* __restrict__ scale_p) {
  int t = blockIdx.x;
  __shared__ float e[GDIM];
  if (threadIdx.x < GDIM) e[threadIdx.x] = expr[(size_t)t * GDIM + threadIdx.x];
  __syncthreads();
  float s = scale_p[0];
#pragma unroll
  for (int q = 0; q < 4; ++q) {
    int d = threadIdx.x + q * 256;
    const float4* gd = (const float4*)(g2d + (size_t)d * GDIM);
    const float4* gu = (const float4*)(g2u + (size_t)d * GDIM);
    const float4* ef = (const float4*)e;
    float sd = 0.0f, su = 0.0f;
#pragma unroll
    for (int i = 0; i < GDIM / 4; ++i) {
      float4 a = gd[i];
      float4 b = gu[i];
      float4 ev = ef[i];
      sd += a.x * ev.x + a.y * ev.y + a.z * ev.z + a.w * ev.w;
      su += b.x * ev.x + b.y * ev.y + b.z * ev.z + b.w * ev.w;
    }
    float dn = tanhf(sd);
    size_t off = (size_t)t * DD + d;
    float xnv = xn[off];
    x[off] += dn * xnv * su * s;
  }
}

// ---------------- host launcher ----------------
extern "C" void kernel_launch(void* const* d_in, const int* in_sizes, int n_in,
                              void* d_out, int out_size, void* d_ws, size_t ws_size,
                              hipStream_t stream) {
  (void)in_sizes; (void)n_in; (void)out_size; (void)ws_size;
  const int*   token_ids = (const int*)d_in[0];
  const float* embed     = (const float*)d_in[1];
  const float* lm_head   = (const float*)d_in[2];
  const float* norm_w    = (const float*)d_in[3];
  const float* genes     = (const float*)d_in[4];
  const float* ctx_w     = (const float*)d_in[5];
  const float* temp      = (const float*)d_in[6];
  const float* g2d_w     = (const float*)d_in[7];
  const float* g2u_w     = (const float*)d_in[8];
  const float* lnorm_w   = (const float*)d_in[9];
  const float* scale     = (const float*)d_in[10];
  float* out = (float*)d_out;

  // workspace layout (~212 MB)
  char* w = (char*)d_ws;
  float*          x      = (float*)w;                              // 8 MB
  float*          xn     = x + (size_t)NTOK * DD;                  // 8 MB
  unsigned short* xnb    = (unsigned short*)(xn + (size_t)NTOK * DD);  // 4 MB
  unsigned short* wbf    = xnb + (size_t)NTOK * DD;                // 64 MB (max V*D bf16)
  float*          logits = (float*)(wbf + (size_t)VV * DD);        // 128 MB
  float*          tv     = logits + (size_t)NTOK * GG;
  int*            ti     = (int*)(tv + (size_t)NTOK * KTOP);
  float*          expr   = (float*)(ti + (size_t)NTOK * KTOP);

  k_embed<<<NTOK, 256, 0, stream>>>(token_ids, embed, x);

  for (int l = 0; l < LLAY; ++l) {
    k_rmsnorm<<<NTOK, 256, 0, stream>>>(x, lnorm_w + (size_t)l * DD, xn, xnb);
    // cast this layer's ctx_w (G x D) to bf16
    {
      int n4 = GG * DD / 4;
      k_cast_bf16<<<n4 / 256, 256, 0, stream>>>(ctx_w + (size_t)l * GG * DD, wbf, n4);
    }
    // logits = (xn @ ctx_w^T) / clip(temp, 0.1)
    k_gemm_bf16<<<dim3(GG / 256, NTOK / 64), 256, 0, stream>>>(
        xnb, wbf, logits, GG, temp + l);
    k_topk<<<NTOK, 32, 0, stream>>>(logits, tv, ti);
    k_express<<<NTOK / 8, 256, 0, stream>>>(tv, ti, genes, expr);
    k_ffn<<<NTOK, 256, 0, stream>>>(x, xn, expr,
                                    g2d_w + (size_t)l * DD * GDIM,
                                    g2u_w + (size_t)l * DD * GDIM,
                                    scale + l);
  }

  // final norm + LM head
  k_rmsnorm<<<NTOK, 256, 0, stream>>>(x, norm_w, xn, xnb);
  {
    int n4 = VV * DD / 4;
    k_cast_bf16<<<n4 / 256, 256, 0, stream>>>(lm_head, wbf, n4);
  }
  k_gemm_bf16<<<dim3(VV / 256, NTOK / 64), 256, 0, stream>>>(
      xnb, wbf, out, VV, (const float*)nullptr);
}
